// RansLMHead_44598940402132
// MI455X (gfx1250) — matrix operand order repaired and checked by reading, not compiled, not run on hardware
//
#include <hip/hip_runtime.h>
#include <hip/hip_bf16.h>
#include <stdint.h>

// ---- types for WMMA fragments (wave32, gfx1250) ----
typedef __attribute__((ext_vector_type(4)))  __bf16 v4bf;
typedef __attribute__((ext_vector_type(8)))  __bf16 v8bf;
typedef __attribute__((ext_vector_type(16))) __bf16 v16bf;
typedef __attribute__((ext_vector_type(8)))  float  v8f;
typedef __attribute__((ext_vector_type(4)))  float  v4f;   // native vector for NT loads

#define M_DIM 256
#define K_DIM 4096
#define N_DIM 32000
#define BN    64      // N columns per block
#define KSTEP 64      // K slice per LDS stage
#define XPAD  72      // padded LDS row stride (bf16 elems): 144B, 16B aligned, conflict-free

// One async chunk: INST_OFFSET is applied to BOTH the LDS dest and the global
// address (ISA 08_async_tensor §4.4), so all 8 chunks share one VGPR address
// pair and one LDS-offset VGPR, differing only in the immediate.
#define ASYNC_X_CHUNK(ofs)                                                  \
  asm volatile("global_load_async_to_lds_b128 %0, %1, off offset:" #ofs     \
               :: "v"(xsRowLds), "v"(g) : "memory")

// ------------------------------------------------------------------
// Kernel 1: one-time x fp32 -> bf16 conversion into workspace (2 MB).
// ------------------------------------------------------------------
__global__ __launch_bounds__(256) void cvt_x_kernel(const float* __restrict__ x,
                                                    __bf16* __restrict__ xb) {
  int i = blockIdx.x * blockDim.x + threadIdx.x;        // handles 4 elems
  const v4f f = ((const v4f*)x)[i];
  v4bf o = { (__bf16)f.x, (__bf16)f.y, (__bf16)f.z, (__bf16)f.w };
  ((v4bf*)xb)[i] = o;
}

// ------------------------------------------------------------------
// Kernel 2: memory-bound LM-head GEMM. Block = 8 waves = M256 x N64 tile.
// W streamed fp32 (non-temporal) -> bf16 once; x tile async-copied to LDS.
// ------------------------------------------------------------------
__global__ __launch_bounds__(256) void lmhead_gemm_kernel(const __bf16* __restrict__ xb,
                                                          const float* __restrict__ W,
                                                          float* __restrict__ out) {
  __shared__ __bf16 xs [M_DIM][XPAD];   // 256*72*2 = 36864 B
  __shared__ __bf16 wsh[BN  ][XPAD];    //  64*72*2 =  9216 B

  const int tid  = threadIdx.x;
  const int lane = tid & 31;
  const int wid  = tid >> 5;
  const int l15  = lane & 15;
  const bool hiH = (lane & 16) != 0;
  const int n0   = blockIdx.x * BN;

  const int wm = wid & 3;      // wave's M sub-block (4 tiles of 16 rows)
  const int wn = wid >> 2;     // wave's N sub-block (2 tiles of 16 cols)

  v8f acc[4][2] = {};

  // per-thread staging addresses (constant across the K loop except +128B/iter)
  const unsigned xsRowLds = (unsigned)(uintptr_t)(&xs[tid][0]); // LDS byte offset (addr[31:0])
  uint64_t g = (uint64_t)(uintptr_t)(xb + tid * K_DIM);         // this thread's x row
  const int    wRow   = tid >> 2;                   // 0..63 local W row
  const int    wq     = tid & 3;                    // 16-float quarter
  const float* wgBase = W + (size_t)(n0 + wRow) * K_DIM + wq * 16;

  for (int k0 = 0; k0 < K_DIM; k0 += KSTEP) {
    // ---- async stage x tile (bf16) straight into LDS: 8 x b128, shared addr ----
    ASYNC_X_CHUNK(0);
    ASYNC_X_CHUNK(16);
    ASYNC_X_CHUNK(32);
    ASYNC_X_CHUNK(48);
    ASYNC_X_CHUNK(64);
    ASYNC_X_CHUNK(80);
    ASYNC_X_CHUNK(96);
    ASYNC_X_CHUNK(112);
    g += KSTEP * 2;  // advance one K slice (bytes)

    // ---- load W slice fp32 (non-temporal stream), convert to bf16, store LDS ----
    {
      const v4f* wp = (const v4f*)(wgBase + k0);
      v4f f0 = __builtin_nontemporal_load(wp + 0);
      v4f f1 = __builtin_nontemporal_load(wp + 1);
      v4f f2 = __builtin_nontemporal_load(wp + 2);
      v4f f3 = __builtin_nontemporal_load(wp + 3);
      v8bf p0 = { (__bf16)f0.x, (__bf16)f0.y, (__bf16)f0.z, (__bf16)f0.w,
                  (__bf16)f1.x, (__bf16)f1.y, (__bf16)f1.z, (__bf16)f1.w };
      v8bf p1 = { (__bf16)f2.x, (__bf16)f2.y, (__bf16)f2.z, (__bf16)f2.w,
                  (__bf16)f3.x, (__bf16)f3.y, (__bf16)f3.z, (__bf16)f3.w };
      *(v8bf*)&wsh[wRow][wq * 16]     = p0;
      *(v8bf*)&wsh[wRow][wq * 16 + 8] = p1;
      if (k0 + KSTEP < K_DIM) __builtin_prefetch(wgBase + k0 + KSTEP, 0, 0);
    }

    asm volatile("s_wait_asynccnt 0" ::: "memory");
    __syncthreads();

    // ---- 2 x (K=32) WMMA steps out of LDS ----
#pragma unroll
    for (int kk = 0; kk < 2; ++kk) {
      // B fragments (32x16 K-major): lanes 0-15 hold K={0..7,16..23}, lanes 16-31 +8
      v16bf bfrag[2];
#pragma unroll
      for (int ni = 0; ni < 2; ++ni) {
        const int wr = (wn * 2 + ni) * 16 + l15;
        const int kb = kk * 32 + (hiH ? 8 : 0);
        v8bf lo = *(const v8bf*)&wsh[wr][kb];
        v8bf h2 = *(const v8bf*)&wsh[wr][kb + 16];
        bfrag[ni] = __builtin_shufflevector(lo, h2,
                      0,1,2,3,4,5,6,7,8,9,10,11,12,13,14,15);
      }
      // A fragments (16x32): lanes 0-15 hold K={0..7,16..23}, lanes 16-31 +8
#pragma unroll
      for (int mi = 0; mi < 4; ++mi) {
        const int xr = (wm * 4 + mi) * 16 + l15;
        const int ka = kk * 32 + (hiH ? 8 : 0);
        v8bf lo = *(const v8bf*)&xs[xr][ka];
        v8bf h2 = *(const v8bf*)&xs[xr][ka + 16];
        v16bf afrag = __builtin_shufflevector(lo, h2,
                        0,1,2,3,4,5,6,7,8,9,10,11,12,13,14,15);
#pragma unroll
        for (int ni = 0; ni < 2; ++ni) {
          acc[mi][ni] = __builtin_amdgcn_wmma_f32_16x16x32_bf16(
              false, afrag, false, bfrag[ni], (short)0, acc[mi][ni], false, false);
        }
      }
    }
    __syncthreads();
  }

  // ---- epilogue: C layout = VGPR r -> M row r (lanes 0-15) / r+8 (lanes 16-31);
  //      lanes 0-15 of each half-wave write 16 consecutive columns (64B bursts).
#pragma unroll
  for (int mi = 0; mi < 4; ++mi) {
#pragma unroll
    for (int ni = 0; ni < 2; ++ni) {
      const int col     = n0 + (wn * 2 + ni) * 16 + l15;
      const int rowBase = (wm * 4 + mi) * 16 + (hiH ? 8 : 0);
#pragma unroll
      for (int r = 0; r < 8; ++r) {
        __builtin_nontemporal_store(acc[mi][ni][r],
                                    out + (size_t)(rowBase + r) * N_DIM + col);
      }
    }
  }
}

extern "C" void kernel_launch(void* const* d_in, const int* in_sizes, int n_in,
                              void* d_out, int out_size, void* d_ws, size_t ws_size,
                              hipStream_t stream) {
  const float* x = (const float*)d_in[0];   // [8,32,4096] fp32
  const float* W = (const float*)d_in[1];   // [32000,4096] fp32
  float* out = (float*)d_out;               // [256,32000] fp32
  __bf16* xb = (__bf16*)d_ws;               // 256*4096*2 = 2 MB scratch

  // 1) x -> bf16 (one pass, 4 elems/thread)
  cvt_x_kernel<<<(M_DIM * K_DIM / 4) / 256, 256, 0, stream>>>(x, xb);
  // 2) GEMM: 500 blocks x 256 threads cover N=32000 with M=256 resident
  lmhead_gemm_kernel<<<N_DIM / BN, 256, 0, stream>>>(xb, W, out);
}
